// BackwardFromTerminal_19138374271109
// MI455X (gfx1250) — matrix operand order repaired and checked by reading, compiled, not run on hardware
//
#include <hip/hip_runtime.h>
#include <hip/hip_bf16.h>
#include <stdint.h>

// ---------------------------------------------------------------------------
// CDNA5 (gfx1250) types for WMMA
// ---------------------------------------------------------------------------
typedef __attribute__((ext_vector_type(16))) __bf16 v16bf;
typedef __attribute__((ext_vector_type(8)))  float  v8f;

// Problem constants (match reference setup_inputs)
#define BB   256   // batch
#define LL   256   // latent dim
#define HH   64    // hidden dim
#define TT   512   // T_out
#define G4H  256   // 4*H

// ---------------------------------------------------------------------------
// WMMA fragment loaders (ISA 7.12.2 layouts, wave32)
// ---------------------------------------------------------------------------
__device__ inline v16bf load_a_frag(const __bf16* a, int ld, int lane) {
  int half = lane >> 4;
  int r    = lane & 15;
  int k0   = half ? 8 : 0;
  v16bf f;
#pragma unroll
  for (int i = 0; i < 4; ++i) {
    f[2 * i]     = a[r * ld + k0 + 2 * i];
    f[2 * i + 1] = a[r * ld + k0 + 2 * i + 1];
  }
#pragma unroll
  for (int i = 0; i < 4; ++i) {
    int k = 16 + k0 + 2 * i;
    f[8 + 2 * i]     = a[r * ld + k];
    f[8 + 2 * i + 1] = a[r * ld + k + 1];
  }
  return f;
}

__device__ inline v16bf load_b_frag(const __bf16* wt, int ldk, int lane) {
  int half = lane >> 4;
  int n    = lane & 15;
  int kb   = half ? 8 : 0;
  v16bf f;
#pragma unroll
  for (int i = 0; i < 8; ++i) {
    int k = kb + 2 * i + (i >= 4 ? 8 : 0);
    f[2 * i]     = wt[n * ldk + k];
    f[2 * i + 1] = wt[n * ldk + k + 1];
  }
  return f;
}

__device__ inline float sigf(float x) { return 1.0f / (1.0f + __expf(-x)); }

// ---------------------------------------------------------------------------
// Generic tiled WMMA GEMM: C(M,N) = A(M,K) @ W(K,N) (+ bias[n])
// fp32 in/out, bf16 WMMA with f32 accumulate.
// Block = 128 threads (4 waves). Tile 64(M) x 64(N) x 32(K).
// Double-buffered LDS staging with vectorized (b128) global loads.
// Requires: M%64==0, N%64==0, K%32==0 (true for every call here).
// ---------------------------------------------------------------------------
#define GM 64
#define GN 64
#define GK 32
#define GKP (GK + 4)   // row byte stride 72 (8B aligned)

__device__ inline void stage_tiles(const float* __restrict__ A,
                                   const float* __restrict__ W,
                                   int mBase, int nBase, int kb, int lda,
                                   int ldw, __bf16 (*aL)[GKP],
                                   __bf16 (*wL)[GKP], int tid) {
  // A tile 64x32: 512 x (4 float) chunks, 4 per thread, b128 loads
#pragma unroll
  for (int q = 0; q < 4; ++q) {
    int c  = q * 128 + tid;
    int r  = c >> 3;          // 8 chunks per row
    int kc = (c & 7) * 4;
    const float4 fv =
        *(const float4*)(A + (size_t)(mBase + r) * lda + kb + kc);
    __bf16 t4[4] = {(__bf16)fv.x, (__bf16)fv.y, (__bf16)fv.z, (__bf16)fv.w};
    uint64_t u;
    __builtin_memcpy(&u, t4, 8);
    *(uint64_t*)(&aL[r][kc]) = u;
  }
  // W tile 32x64 -> transposed [n][k]: b128 loads, 4 b16 scatter stores
#pragma unroll
  for (int q = 0; q < 4; ++q) {
    int c  = q * 128 + tid;
    int k  = c >> 4;          // 16 chunks per k-row
    int n0 = (c & 15) * 4;
    const float4 fv =
        *(const float4*)(W + (size_t)(kb + k) * ldw + nBase + n0);
    wL[n0 + 0][k] = (__bf16)fv.x;
    wL[n0 + 1][k] = (__bf16)fv.y;
    wL[n0 + 2][k] = (__bf16)fv.z;
    wL[n0 + 3][k] = (__bf16)fv.w;
  }
}

__global__ __launch_bounds__(128) void gemm_wmma(
    const float* __restrict__ A, const float* __restrict__ W,
    const float* __restrict__ bias, float* __restrict__ C,
    int M, int N, int K, int lda, int ldw, int ldc) {
  (void)M; (void)N;
  __shared__ __bf16 aLds[2][GM][GKP];
  __shared__ __bf16 wLds[2][GN][GKP];

  const int tid  = threadIdx.x;
  const int wave = tid >> 5;
  const int lane = tid & 31;
  const int mBase = blockIdx.y * GM;
  const int nBase = blockIdx.x * GN;
  const int nK = K >> 5;

  v8f acc[4];
#pragma unroll
  for (int t = 0; t < 4; ++t)
#pragma unroll
    for (int v = 0; v < 8; ++v) acc[t][v] = 0.0f;

  // prologue: stage K-slice 0 into buffer 0
  stage_tiles(A, W, mBase, nBase, 0, lda, ldw, aLds[0], wLds[0], tid);

  for (int ki = 0; ki < nK; ++ki) {
    __syncthreads();  // staged data for slice ki is visible
    // software pipeline: stage slice ki+1 into the other buffer
    if (ki + 1 < nK)
      stage_tiles(A, W, mBase, nBase, (ki + 1) << 5, lda, ldw,
                  aLds[(ki + 1) & 1], wLds[(ki + 1) & 1], tid);

    const int b = ki & 1;
    v16bf af = load_a_frag(&aLds[b][wave * 16][0], GKP, lane);
#pragma unroll
    for (int nt = 0; nt < 4; ++nt) {
      v16bf bf = load_b_frag(&wLds[b][nt * 16][0], GKP, lane);
      acc[nt] = __builtin_amdgcn_wmma_f32_16x16x32_bf16(
          false, af, false, bf, (short)0, acc[nt], false, false);
    }
  }

  // store: C/D layout -> col = lane&15, row = v + 8*(lane>>4)
  const int col0 = lane & 15;
  const int rsel = lane >> 4;
#pragma unroll
  for (int nt = 0; nt < 4; ++nt) {
    int col  = nBase + nt * 16 + col0;
    float bv = bias ? bias[col] : 0.0f;
#pragma unroll
    for (int v = 0; v < 8; ++v) {
      int row = mBase + wave * 16 + v + 8 * rsel;
      C[(size_t)row * ldc + col] = acc[nt][v] + bv;
    }
  }
}

// ---------------------------------------------------------------------------
// Expand: xproj[b,t,n] = zbase[b,n] + pos(t)*w_proj[L, n] + b_proj[n]
// ---------------------------------------------------------------------------
__global__ __launch_bounds__(256) void expand_x(
    const float* __restrict__ zbase, const float* __restrict__ wlast,
    const float* __restrict__ bproj, float* __restrict__ xproj) {
  size_t gid = (size_t)blockIdx.x * 256 + threadIdx.x;
  int n = (int)(gid & 255);
  size_t bt = gid >> 8;
  int t = (int)(bt % TT);
  int b = (int)(bt / TT);
  float pos = (float)t * (1.0f / (float)(TT - 1));
  xproj[gid] = zbase[(size_t)b * LL + n] + pos * wlast[n] + bproj[n];
}

// ---------------------------------------------------------------------------
// LSTM scan. One block owns BT=16 batch rows and iterates all T steps with
// Wh resident in LDS+VGPRs, h/c resident in LDS. Per step:
//   gates = XG[:,t,:] + h @ Wh   (WMMA, M=16, K=64, N=256)
//   c = sig(f)*c + sig(i)*tanh(g);  h = sig(o)*tanh(c)
// XG tiles are double-buffered in LDS and prefetched one step ahead with
// GLOBAL_LOAD_ASYNC_TO_LDS_B128 (ASYNCcnt), hiding global latency.
// ---------------------------------------------------------------------------
#define BT 16
#define WHP 72   // padded k-stride for whT / hB
#define GATEP 260

__device__ inline void async_stage_xg(const float* __restrict__ XG, int b0,
                                      int tstep, float* ldsRowBase, int tid) {
  // 16 rows x 256 floats = 1024 x 16B chunks; 8 wave-wide B128 async loads
#pragma unroll
  for (int q = 0; q < 8; ++q) {
    int c  = q * 128 + tid;
    int r  = c >> 6;          // row 0..15 (64 chunks per row)
    int n0 = (c & 63) * 4;    // float offset within row
    uint64_t ga = (uint64_t)(uintptr_t)(XG +
        ((size_t)(b0 + r) * TT + tstep) * G4H + n0);
    // low 32 bits of a generic pointer into LDS == LDS byte offset
    uint32_t lo = (uint32_t)(uintptr_t)(ldsRowBase + (size_t)r * GATEP + n0);
    asm volatile("global_load_async_to_lds_b128 %0, %1, off"
                 :: "v"(lo), "v"(ga)
                 : "memory");
  }
}

__global__ __launch_bounds__(128) void lstm_scan(
    const float* __restrict__ XG, const float* __restrict__ Wh,
    float* __restrict__ out, int outStride, int outOff, int reverse) {
  __shared__ __bf16 whT[G4H][WHP];       // [n][k] = Wh[k][n], 36.9 KB
  __shared__ __bf16 hB[BT][WHP];         // current h (bf16)
  __shared__ float  gate[2][BT][GATEP];  // double-buffered z tiles, 33.3 KB
  __shared__ float  cS[BT][HH];          // cell state

  const int tid  = threadIdx.x;
  const int wave = tid >> 5;
  const int lane = tid & 31;
  const int b0   = blockIdx.x * BT;

  // load Wh (64 x 256) transposed into LDS, coalesced over n (128 iters)
#pragma unroll 4
  for (int q = 0; q < (HH * G4H) / 128; ++q) {
    int idx = q * 128 + tid;
    int k = idx >> 8, n = idx & 255;
    whT[n][k] = (__bf16)Wh[(size_t)k * G4H + n];
  }
  // zero h, c (8 iters)
#pragma unroll
  for (int q = 0; q < (BT * HH) / 128; ++q) {
    int idx = q * 128 + tid;
    int r = idx >> 6, j = idx & 63;
    hB[r][j] = (__bf16)0.0f;
    cS[r][j] = 0.0f;
  }

  // prologue: async-stage xg(t=0) into buffer 0, retire, then sync staging
  {
    int tt0 = reverse ? (TT - 1) : 0;
    async_stage_xg(XG, b0, tt0, &gate[0][0][0], tid);
    asm volatile("s_wait_asynccnt 0" ::: "memory");
  }
  __syncthreads();

  // hoist loop-invariant B fragments (Wh) into VGPRs: 4 N-tiles x 2 K-halves
  v16bf bfr[4][2];
#pragma unroll
  for (int nt = 0; nt < 4; ++nt) {
    int nbase = wave * 64 + nt * 16;
    bfr[nt][0] = load_b_frag(&whT[nbase][0], WHP, lane);
    bfr[nt][1] = load_b_frag(&whT[nbase][32], WHP, lane);
  }

  const int half = lane >> 4;
  const int col0 = lane & 15;

  for (int t = 0; t < TT; ++t) {
    const int tt   = reverse ? (TT - 1 - t) : t;
    const int buf  = t & 1;
    const int nbuf = buf ^ 1;

    // prefetch xg(t+1) into the other buffer (clamped dummy on last step)
    {
      int tnext = (t + 1 < TT) ? (t + 1) : t;
      int ttn   = reverse ? (TT - 1 - tnext) : tnext;
      async_stage_xg(XG, b0, ttn, &gate[nbuf][0][0], tid);
    }
    // retire this step's 8 async loads (loads complete in order), keep the
    // 8 just-issued prefetches in flight
    asm volatile("s_wait_asynccnt 8" ::: "memory");
    __syncthreads();

    // gate[buf] += h @ Wh : each wave covers 64 columns (4 tiles), K=64
    {
      v16bf a0 = load_a_frag(&hB[0][0], WHP, lane);
      v16bf a1 = load_a_frag(&hB[0][32], WHP, lane);
#pragma unroll
      for (int nt = 0; nt < 4; ++nt) {
        v8f acc;
#pragma unroll
        for (int v = 0; v < 8; ++v) acc[v] = 0.0f;
        acc = __builtin_amdgcn_wmma_f32_16x16x32_bf16(
            false, a0, false, bfr[nt][0], (short)0, acc, false, false);
        acc = __builtin_amdgcn_wmma_f32_16x16x32_bf16(
            false, a1, false, bfr[nt][1], (short)0, acc, false, false);
        int n = wave * 64 + nt * 16 + col0;
#pragma unroll
        for (int v = 0; v < 8; ++v) gate[buf][v + 8 * half][n] += acc[v];
      }
    }
    __syncthreads();

    // gate nonlinearity + state update; write h out (8 iters, static)
#pragma unroll
    for (int q = 0; q < (BT * HH) / 128; ++q) {
      int idx = q * 128 + tid;
      int r = idx >> 6, j = idx & 63;
      float zi = gate[buf][r][j];
      float zf = gate[buf][r][HH + j];
      float zg = gate[buf][r][2 * HH + j];
      float zo = gate[buf][r][3 * HH + j];
      float c = cS[r][j];
      c = sigf(zf) * c + sigf(zi) * tanhf(zg);
      float h = sigf(zo) * tanhf(c);
      cS[r][j] = c;
      hB[r][j] = (__bf16)h;
      out[((size_t)(b0 + r) * TT + tt) * outStride + outOff + j] = h;
    }
    __syncthreads();  // protect gate[nbuf] (async target) and hB
  }
}

// ---------------------------------------------------------------------------
// LayerNorm (over 128 features) + tanh-approx GELU, in place. Wave per row.
// ---------------------------------------------------------------------------
__global__ __launch_bounds__(128) void ln_gelu(
    float* __restrict__ h, const float* __restrict__ sc,
    const float* __restrict__ bi) {
  int row  = blockIdx.x * 4 + (threadIdx.x >> 5);
  int lane = threadIdx.x & 31;
  float* p = h + (size_t)row * 128;

  float v[4], s = 0.0f, s2 = 0.0f;
#pragma unroll
  for (int k = 0; k < 4; ++k) {
    v[k] = p[k * 32 + lane];
    s += v[k];
    s2 += v[k] * v[k];
  }
#pragma unroll
  for (int off = 16; off > 0; off >>= 1) {
    s  += __shfl_xor(s, off, 32);
    s2 += __shfl_xor(s2, off, 32);
  }
  float mu   = s * (1.0f / 128.0f);
  float var  = s2 * (1.0f / 128.0f) - mu * mu;
  float rstd = rsqrtf(var + 1e-6f);
#pragma unroll
  for (int k = 0; k < 4; ++k) {
    int j   = k * 32 + lane;
    float x = (v[k] - mu) * rstd * sc[j] + bi[j];
    float g = 0.5f * x *
              (1.0f + tanhf(0.7978845608f * (x + 0.044715f * x * x * x)));
    p[j] = g;
  }
}

// ---------------------------------------------------------------------------
// Host-side orchestration
// ---------------------------------------------------------------------------
extern "C" void kernel_launch(void* const* d_in, const int* in_sizes, int n_in,
                              void* d_out, int out_size, void* d_ws,
                              size_t ws_size, hipStream_t stream) {
  (void)in_sizes; (void)n_in; (void)out_size; (void)ws_size;

  const float* z_T      = (const float*)d_in[0];
  const float* w_proj   = (const float*)d_in[1];   // (257,256)
  const float* b_proj   = (const float*)d_in[2];
  const float* e0f_Wx   = (const float*)d_in[3];
  const float* e0f_Wh   = (const float*)d_in[4];
  const float* e0f_b    = (const float*)d_in[5];
  const float* e0b_Wx   = (const float*)d_in[6];
  const float* e0b_Wh   = (const float*)d_in[7];
  const float* e0b_b    = (const float*)d_in[8];
  const float* e1f_Wx   = (const float*)d_in[9];
  const float* e1f_Wh   = (const float*)d_in[10];
  const float* e1f_b    = (const float*)d_in[11];
  const float* e1b_Wx   = (const float*)d_in[12];
  const float* e1b_Wh   = (const float*)d_in[13];
  const float* e1b_b    = (const float*)d_in[14];
  const float* d_Wx     = (const float*)d_in[15];
  const float* d_Wh     = (const float*)d_in[16];
  const float* d_b      = (const float*)d_in[17];
  const float* w_mlp1   = (const float*)d_in[18];
  const float* b_mlp1   = (const float*)d_in[19];
  const float* ln_scale = (const float*)d_in[20];
  const float* ln_bias  = (const float*)d_in[21];
  const float* w_out    = (const float*)d_in[22];
  const float* b_out    = (const float*)d_in[23];

  const size_t MBT = (size_t)BB * TT;            // 131072 rows

  // workspace carve-out (bytes)
  char* ws = (char*)d_ws;
  float* xproj = (float*)(ws);                                     // 128 MB
  float* XG    = (float*)(ws + MBT * 256 * 4);                     // 128 MB
  float* h0    = (float*)(ws + 2 * MBT * 256 * 4);                 // 64 MB
  float* h1    = (float*)(ws + 2 * MBT * 256 * 4 + MBT * 128 * 4); // 64 MB
  float* zbase = (float*)(ws + 2 * MBT * 256 * 4 + 2 * MBT * 128 * 4);
  float* dec   = h0;      // h0 dead once XG for e1b consumed
  float* mlph  = xproj;   // xproj dead once XG for e0b consumed

  const dim3 blk(128);
  const dim3 gBig4(4, (unsigned)(MBT / GM));   // N=256 GEMMs over B*T rows
  const dim3 gBig2(2, (unsigned)(MBT / GM));   // N=128
  const dim3 gScan(BB / BT);

  // 1) zbase = z_T @ w_proj[0:256,:]
  gemm_wmma<<<dim3(4, 4), blk, 0, stream>>>(z_T, w_proj, nullptr, zbase,
                                            BB, LL, LL, LL, LL, LL);
  // 2) xproj[b,t,:] = zbase[b,:] + pos(t)*w_proj[256,:] + b_proj
  expand_x<<<dim3((unsigned)MBT), dim3(256), 0, stream>>>(
      zbase, w_proj + (size_t)LL * LL, b_proj, xproj);

  // 3) encoder layer 0
  gemm_wmma<<<gBig4, blk, 0, stream>>>(xproj, e0f_Wx, e0f_b, XG,
                                       (int)MBT, G4H, LL, LL, G4H, G4H);
  lstm_scan<<<gScan, blk, 0, stream>>>(XG, e0f_Wh, h0, 128, 0, 0);
  gemm_wmma<<<gBig4, blk, 0, stream>>>(xproj, e0b_Wx, e0b_b, XG,
                                       (int)MBT, G4H, LL, LL, G4H, G4H);
  lstm_scan<<<gScan, blk, 0, stream>>>(XG, e0b_Wh, h0, 128, 64, 1);

  // 4) encoder layer 1 (input 2H=128)
  gemm_wmma<<<gBig4, blk, 0, stream>>>(h0, e1f_Wx, e1f_b, XG,
                                       (int)MBT, G4H, 128, 128, G4H, G4H);
  lstm_scan<<<gScan, blk, 0, stream>>>(XG, e1f_Wh, h1, 128, 0, 0);
  gemm_wmma<<<gBig4, blk, 0, stream>>>(h0, e1b_Wx, e1b_b, XG,
                                       (int)MBT, G4H, 128, 128, G4H, G4H);
  lstm_scan<<<gScan, blk, 0, stream>>>(XG, e1b_Wh, h1, 128, 64, 1);

  // 5) decoder LSTM (input 2H=128, output H=64)
  gemm_wmma<<<gBig4, blk, 0, stream>>>(h1, d_Wx, d_b, XG,
                                       (int)MBT, G4H, 128, 128, G4H, G4H);
  lstm_scan<<<gScan, blk, 0, stream>>>(XG, d_Wh, dec, 64, 0, 0);

  // 6) MLP head
  gemm_wmma<<<gBig2, blk, 0, stream>>>(dec, w_mlp1, b_mlp1, mlph,
                                       (int)MBT, 128, HH, HH, 128, 128);
  ln_gelu<<<dim3((unsigned)(MBT / 4)), blk, 0, stream>>>(mlph, ln_scale,
                                                         ln_bias);
  gemm_wmma<<<gBig4, blk, 0, stream>>>(mlph, w_out, b_out, (float*)d_out,
                                       (int)MBT, LL, 128, 128, LL, LL);
}
